// TemporalAttention_51531017617994
// MI455X (gfx1250) — matrix operand order repaired and checked by reading, not compiled, not run on hardware
//
#include <hip/hip_runtime.h>
#include <hip/hip_bf16.h>

typedef __attribute__((ext_vector_type(16))) __bf16 v16bf;
typedef __attribute__((ext_vector_type(8)))  __bf16 v8bf;
typedef __attribute__((ext_vector_type(2)))  __bf16 v2bf;
typedef __attribute__((ext_vector_type(8)))  float  v8f;

#define B_    4
#define T_    8
#define C_    128
#define D_    64
#define HW_   (96*96)
#define PIX_  16

// Strides chosen so px-stride == 4 (mod 64 banks): 16 pixels x 4-bank b128
// accesses tile all 64 banks exactly (conflict-free) AND stay 16B-aligned.
#define QK_STRIDE 516   // T_*D_ + 4 floats
#define V_STRIDE  1028  // T_*C_ + 4 floats
#define S_STRIDE  68    // T_*T_ + 4 floats (16B-aligned rows)
#define X_STRIDE  136   // C_ + 8 bf16 elements (272 B, 16B aligned)

// Offsets inside the fused f32 LDS arena (floats)
#define QS_BASE 0
#define KS_BASE (PIX_ * QK_STRIDE)                 // 8256
#define VS_BASE (2 * PIX_ * QK_STRIDE)             // 16512
#define SC_BASE (2 * PIX_ * QK_STRIDE + PIX_ * V_STRIDE)   // 32960
#define SMEM_F  (SC_BASE + PIX_ * S_STRIDE)        // 34048 floats

__global__ __launch_bounds__(256)
void temporal_attn_kernel(const float* __restrict__ x,
                          const float* __restrict__ Wq, const float* __restrict__ bq,
                          const float* __restrict__ Wk, const float* __restrict__ bk,
                          const float* __restrict__ Wv, const float* __restrict__ bv,
                          float* __restrict__ out)
{
    __shared__ __align__(16) float  smem[SMEM_F];
    __shared__ __align__(16) __bf16 Xt[PIX_ * X_STRIDE];

    const int tid  = threadIdx.x;
    const int lane = tid & 31;
    const int wave = tid >> 5;

    const int bIdx = blockIdx.x / (HW_ / PIX_);
    const int pix0 = (blockIdx.x % (HW_ / PIX_)) * PIX_;

    // ---- Per-wave weight A-fragments (bf16) + bias regs, reused for all t --
    // Jobs 0-3: Q m-tiles, 4-7: K m-tiles, 8-15: V m-tiles. Wave w owns 2w,2w+1.
    v16bf afr[2][4];
    v8f   biasr[2];
    int   dstOff[2], m0j[2], strj[2], rowj[2];

    const int aRow  = lane & 15;           // M within tile (both lane halves)
    const int aOff1 = (lane < 16) ? 0 : 8; // K sub-offset per 16-bit A-layout
    const int dHi   = (lane >= 16) ? 8 : 0;

    #pragma unroll
    for (int jj = 0; jj < 2; ++jj) {
        const int job = wave * 2 + jj;
        const float* W;
        const float* bArr;
        if (job < 4)       { W = Wq; bArr = bq; m0j[jj] = job * 16;       dstOff[jj] = QS_BASE; strj[jj] = QK_STRIDE; rowj[jj] = D_; }
        else if (job < 8)  { W = Wk; bArr = bk; m0j[jj] = (job - 4) * 16; dstOff[jj] = KS_BASE; strj[jj] = QK_STRIDE; rowj[jj] = D_; }
        else               { W = Wv; bArr = bv; m0j[jj] = (job - 8) * 16; dstOff[jj] = VS_BASE; strj[jj] = V_STRIDE;  rowj[jj] = C_; }

        const float4 bb0 = *(const float4*)(bArr + m0j[jj] + dHi);
        const float4 bb1 = *(const float4*)(bArr + m0j[jj] + dHi + 4);
        biasr[jj][0]=bb0.x; biasr[jj][1]=bb0.y; biasr[jj][2]=bb0.z; biasr[jj][3]=bb0.w;
        biasr[jj][4]=bb1.x; biasr[jj][5]=bb1.y; biasr[jj][6]=bb1.z; biasr[jj][7]=bb1.w;

        const int row = m0j[jj] + aRow;
        #pragma unroll
        for (int kf = 0; kf < 4; ++kf) {
            const float* wp = W + row * C_ + kf * 32 + aOff1;
            const float4 f0 = *(const float4*)(wp +  0);
            const float4 f1 = *(const float4*)(wp +  4);
            const float4 f2 = *(const float4*)(wp + 16);
            const float4 f3 = *(const float4*)(wp + 20);
            v16bf a;
            a[0]=(__bf16)f0.x; a[1]=(__bf16)f0.y; a[2]=(__bf16)f0.z;  a[3]=(__bf16)f0.w;
            a[4]=(__bf16)f1.x; a[5]=(__bf16)f1.y; a[6]=(__bf16)f1.z;  a[7]=(__bf16)f1.w;
            a[8]=(__bf16)f2.x; a[9]=(__bf16)f2.y; a[10]=(__bf16)f2.z; a[11]=(__bf16)f2.w;
            a[12]=(__bf16)f3.x;a[13]=(__bf16)f3.y;a[14]=(__bf16)f3.z; a[15]=(__bf16)f3.w;
            afr[jj][kf] = a;
        }
    }

    // ---- Phase 1: projections, per frame t --------------------------------
    const int bN   = lane & 15;             // B-fragment column (pixel)
    const int bKof = (lane < 16) ? 0 : 8;   // B-fragment K sub-offset

    // staging map: thread owns 2 consecutive channels x 4 consecutive pixels
    const int sPg = tid & 3;                // pixel group: px0 = 4*sPg
    const int sC0 = (tid >> 2) * 2;         // channel pair base

    for (int t = 0; t < T_; ++t) {
        __syncthreads();   // previous frame's Xt fully consumed
        {   // stage x[b,t,:,tile] -> bf16 LDS; packed-pair b32 stores
            const float* xp = x + (((size_t)bIdx * T_ + t) * C_ + sC0) * HW_
                            + pix0 + sPg * 4;
            const float4 r0 = *(const float4*)(xp);         // row c0,   4 px
            const float4 r1 = *(const float4*)(xp + HW_);   // row c0+1, 4 px
            const float c0v[4] = {r0.x, r0.y, r0.z, r0.w};
            const float c1v[4] = {r1.x, r1.y, r1.z, r1.w};
            #pragma unroll
            for (int i = 0; i < 4; ++i) {
                v2bf p; p[0] = (__bf16)c0v[i]; p[1] = (__bf16)c1v[i];
                *(v2bf*)&Xt[(sPg * 4 + i) * X_STRIDE + sC0] = p;
            }
            if (t + 1 < T_)  // next frame, same channels/pixels
                __builtin_prefetch(xp + (size_t)C_ * HW_, 0, 1);
        }
        __syncthreads();

        // B-fragments (X tile, K=c x N=pixel) from LDS
        v16bf bfr[4];
        #pragma unroll
        for (int kf = 0; kf < 4; ++kf) {
            const __bf16* p = &Xt[bN * X_STRIDE + kf * 32 + bKof];
            union { v16bf v; v8bf h[2]; } u;
            u.h[0] = *(const v8bf*)(p);
            u.h[1] = *(const v8bf*)(p + 16);
            bfr[kf] = u.v;
        }

        #pragma unroll
        for (int jj = 0; jj < 2; ++jj) {
            v8f acc = {0.f, 0.f, 0.f, 0.f, 0.f, 0.f, 0.f, 0.f};
            #pragma unroll
            for (int kf = 0; kf < 4; ++kf)
                acc = __builtin_amdgcn_wmma_f32_16x16x32_bf16(
                          false, afr[jj][kf], false, bfr[kf],
                          (short)0, acc, false, false);
            acc += biasr[jj];

            // 8 contiguous floats per lane -> two ds_store_b128 (16B aligned)
            const int addr = dstOff[jj] + (lane & 15) * strj[jj]
                           + t * rowj[jj] + m0j[jj] + dHi;
            *(float4*)&smem[addr]     = make_float4(acc[0], acc[1], acc[2], acc[3]);
            *(float4*)&smem[addr + 4] = make_float4(acc[4], acc[5], acc[6], acc[7]);
        }
    }
    __syncthreads();

    // ---- Phase 2: per-pixel T=8 attention in f32 --------------------------
    const int px = tid & 15;
    const int j  = tid >> 4;   // 16 workers per pixel

    // scores: worker j owns Q row tq = j>>1 (kept in regs) x 4 K rows
    {
        const int tq = j >> 1;
        float qreg[D_];
        const float* qp = &smem[QS_BASE + px * QK_STRIDE + tq * D_];
        #pragma unroll
        for (int d4 = 0; d4 < D_ / 4; ++d4)
            *(float4*)&qreg[d4 * 4] = *(const float4*)(qp + d4 * 4);

        #pragma unroll
        for (int q = 0; q < 4; ++q) {
            const int sq = (j & 1) * 4 + q;
            const float* kp = &smem[KS_BASE + px * QK_STRIDE + sq * D_];
            float acc = 0.f;
            #pragma unroll
            for (int d = 0; d < D_; ++d) acc += qreg[d] * kp[d];
            smem[SC_BASE + px * S_STRIDE + tq * 8 + sq] = acc * 0.125f; // 1/sqrt(64)
        }
    }
    __syncthreads();

    // softmax: workers 0..7 each own one t-row of 8 scores
    if (j < 8) {
        float* row = &smem[SC_BASE + px * S_STRIDE + j * 8];
        float m = row[0];
        #pragma unroll
        for (int s = 1; s < 8; ++s) m = fmaxf(m, row[s]);
        float e[8], sum = 0.f;
        #pragma unroll
        for (int s = 0; s < 8; ++s) { e[s] = __expf(row[s] - m); sum += e[s]; }
        const float inv = 1.f / sum;
        #pragma unroll
        for (int s = 0; s < 8; ++s) row[s] = e[s] * inv;
    }
    __syncthreads();

    // out = attn @ V: hoist this thread's V panel (8s x 8e) into registers,
    // then all 8 frames run register-resident FMAs.
    const int e0 = j * 8;
    float vregs[8][8];   // [s][ee]
    #pragma unroll
    for (int s = 0; s < 8; ++s) {
        const float* vp = &smem[VS_BASE + px * V_STRIDE + s * C_ + e0];
        *(float4*)&vregs[s][0] = *(const float4*)(vp);
        *(float4*)&vregs[s][4] = *(const float4*)(vp + 4);
    }

    const size_t outBase = (size_t)bIdx * T_ * C_ * HW_ + pix0 + px;
    for (int t = 0; t < T_; ++t) {
        const float* ar = &smem[SC_BASE + px * S_STRIDE + t * 8];
        float a[8];
        *(float4*)&a[0] = *(const float4*)(ar);
        *(float4*)&a[4] = *(const float4*)(ar + 4);
        float o[8] = {0.f, 0.f, 0.f, 0.f, 0.f, 0.f, 0.f, 0.f};
        #pragma unroll
        for (int s = 0; s < 8; ++s)
            #pragma unroll
            for (int ee = 0; ee < 8; ++ee)
                o[ee] += a[s] * vregs[s][ee];
        #pragma unroll
        for (int ee = 0; ee < 8; ++ee)
            out[outBase + ((size_t)t * C_ + e0 + ee) * HW_] = o[ee];
    }
}

extern "C" void kernel_launch(void* const* d_in, const int* in_sizes, int n_in,
                              void* d_out, int out_size, void* d_ws, size_t ws_size,
                              hipStream_t stream) {
    const float* x  = (const float*)d_in[0];
    const float* Wq = (const float*)d_in[1];
    const float* bq = (const float*)d_in[2];
    const float* Wk = (const float*)d_in[3];
    const float* bk = (const float*)d_in[4];
    const float* Wv = (const float*)d_in[5];
    const float* bv = (const float*)d_in[6];
    float* out = (float*)d_out;

    dim3 grid(B_ * (HW_ / PIX_));   // 4 * 576 = 2304 workgroups
    dim3 block(256);                // 8 wave32s
    temporal_attn_kernel<<<grid, block, 0, stream>>>(x, Wq, bq, Wk, bk, Wv, bv, out);
}
